// MobileBlock_90452011254294
// MI455X (gfx1250) — matrix operand order repaired and checked by reading, not compiled, not run on hardware
//
#include <hip/hip_runtime.h>

// Problem shape (fixed by reference): x [1,256,224,224] f32
// depthwise 3x3 pad1 + BN + ReLU -> pointwise 1x1 (256->512) + BN + ReLU
#define W_IMG   224
#define H_IMG   224
#define HW_IN   (W_IMG * H_IMG)     // 50176
#define C_IN    256
#define C_OUT   512

typedef __attribute__((ext_vector_type(16))) __bf16 bf16x16;
typedef __attribute__((ext_vector_type(8)))  float  floatx8;

union FragB16 {
    uint4   q[2];   // 2 x 16B = 16 bf16
    bf16x16 b;
};

__device__ __forceinline__ unsigned short f32_to_bf16_rne(float f) {
    unsigned int u = __float_as_uint(f);
    u += 0x7FFFu + ((u >> 16) & 1u);     // round-to-nearest-even
    return (unsigned short)(u >> 16);
}

// ---------------------------------------------------------------------------
// Kernel 1: fold BN params into scale/shift, convert pointwise weights to bf16
// ---------------------------------------------------------------------------
__global__ __launch_bounds__(256) void prep_kernel(
    const float* __restrict__ pw_w,
    const float* __restrict__ dw_gamma, const float* __restrict__ dw_beta,
    const float* __restrict__ dw_mean,  const float* __restrict__ dw_var,
    const float* __restrict__ pw_gamma, const float* __restrict__ pw_beta,
    const float* __restrict__ pw_mean,  const float* __restrict__ pw_var,
    unsigned short* __restrict__ wb,     // [C_OUT, C_IN] bf16
    float* __restrict__ dw_scale, float* __restrict__ dw_shift,
    float* __restrict__ pw_scale, float* __restrict__ pw_shift)
{
    int idx = blockIdx.x * 256 + threadIdx.x;
    if (idx < C_OUT * C_IN) wb[idx] = f32_to_bf16_rne(pw_w[idx]);
    if (idx < C_IN) {
        float s = dw_gamma[idx] * rsqrtf(dw_var[idx] + 1e-5f);
        dw_scale[idx] = s;
        dw_shift[idx] = dw_beta[idx] - dw_mean[idx] * s;
    }
    if (idx < C_OUT) {
        float s = pw_gamma[idx] * rsqrtf(pw_var[idx] + 1e-5f);
        pw_scale[idx] = s;
        pw_shift[idx] = pw_beta[idx] - pw_mean[idx] * s;
    }
}

// ---------------------------------------------------------------------------
// Kernel 2: depthwise 3x3 + BN + ReLU; write y bf16 in [HW, C] layout (B^T)
// Block = one (channel, row): lanes along w -> coalesced f32 reads.
// ---------------------------------------------------------------------------
__global__ __launch_bounds__(224) void dw_kernel(
    const float* __restrict__ x,         // [C_IN, H, W]
    const float* __restrict__ dww,       // [C_IN, 1, 3, 3]
    const float* __restrict__ scale, const float* __restrict__ shift,
    unsigned short* __restrict__ y)      // [HW, C_IN] bf16
{
    const int h = blockIdx.x;
    const int c = blockIdx.y;
    const int w = threadIdx.x;

    float k[9];
#pragma unroll
    for (int i = 0; i < 9; ++i) k[i] = dww[c * 9 + i];

    const float* xp = x + (size_t)c * HW_IN;
    float s = 0.f;
#pragma unroll
    for (int dh = -1; dh <= 1; ++dh) {
        int hh = h + dh;
        if (hh < 0 || hh >= H_IMG) continue;
        const float* row = xp + hh * W_IMG;
#pragma unroll
        for (int dw2 = -1; dw2 <= 1; ++dw2) {
            int ww = w + dw2;
            if (ww < 0 || ww >= W_IMG) continue;
            s += k[(dh + 1) * 3 + (dw2 + 1)] * row[ww];
        }
    }
    float v = s * scale[c] + shift[c];
    v = v > 0.f ? v : 0.f;
    y[(size_t)(h * W_IMG + w) * C_IN + c] = f32_to_bf16_rne(v);
}

// ---------------------------------------------------------------------------
// Kernel 3: pointwise GEMM z[M=512, N=50176] = A[512,256] x y[256,50176]
// via v_wmma_f32_16x16x32_bf16, fused BN + ReLU epilogue.
// Block: 256 threads = 8 waves, tile 128(M) x 128(N); wave: 64(M) x 32(N).
// ---------------------------------------------------------------------------
__global__ __launch_bounds__(256) void pw_gemm_kernel(
    const unsigned short* __restrict__ A,   // [512, 256] bf16 row-major
    const unsigned short* __restrict__ Y,   // [50176, 256] bf16 (= B^T)
    const float* __restrict__ scale, const float* __restrict__ shift,
    float* __restrict__ out)                // [512, 50176] f32
{
    const int K    = C_IN;
    const int tid  = threadIdx.x;
    const int wave = tid >> 5;
    const int lane = tid & 31;
    const int half = lane >> 4;   // 0: lanes 0-15, 1: lanes 16-31
    const int l16  = lane & 15;
    const int mw   = wave >> 2;   // 0..1
    const int nw   = wave & 3;    // 0..3
    const int m_base = blockIdx.y * 128 + mw * 64;
    const int n_base = blockIdx.x * 128 + nw * 32;

    floatx8 acc[4][2];
    const floatx8 zero = {0.f, 0.f, 0.f, 0.f, 0.f, 0.f, 0.f, 0.f};
#pragma unroll
    for (int mi = 0; mi < 4; ++mi)
#pragma unroll
        for (int ni = 0; ni < 2; ++ni) acc[mi][ni] = zero;

    const unsigned short* Ap[4];
#pragma unroll
    for (int mi = 0; mi < 4; ++mi)
        Ap[mi] = A + (size_t)(m_base + mi * 16 + l16) * K;
    const unsigned short* Bp[2];
#pragma unroll
    for (int ni = 0; ni < 2; ++ni)
        Bp[ni] = Y + (size_t)(n_base + ni * 16 + l16) * K;

    for (int k0 = 0; k0 < K; k0 += 32) {
        if (k0 + 32 < K) {
            __builtin_prefetch(Bp[0] + k0 + 32, 0, 1);   // global_prefetch_b8
            __builtin_prefetch(Bp[1] + k0 + 32, 0, 1);
        }
        FragB16 fa[4], fb[2];
        // A 16x32 bf16 layout: lanes 0-15 hold K k0+[0..7] (v0..3) and
        // k0+16+[0..7] (v4..7); lanes 16-31 hold k0+8+[0..7] / k0+24+[0..7].
#pragma unroll
        for (int mi = 0; mi < 4; ++mi) {
            const unsigned short* p = Ap[mi] + k0 + half * 8;
            fa[mi].q[0] = *reinterpret_cast<const uint4*>(p);
            fa[mi].q[1] = *reinterpret_cast<const uint4*>(p + 16);
        }
        // B 32x16 bf16 layout: lane = column N; lanes 0-15 hold K k0+[0..7] and
        // k0+16+[0..7]; lanes 16-31 hold k0+8+[0..7] / k0+24+[0..7] (same as A).
#pragma unroll
        for (int ni = 0; ni < 2; ++ni) {
            const unsigned short* p = Bp[ni] + k0 + half * 8;
            fb[ni].q[0] = *reinterpret_cast<const uint4*>(p);
            fb[ni].q[1] = *reinterpret_cast<const uint4*>(p + 16);
        }
#pragma unroll
        for (int mi = 0; mi < 4; ++mi)
#pragma unroll
            for (int ni = 0; ni < 2; ++ni)
                acc[mi][ni] = __builtin_amdgcn_wmma_f32_16x16x32_bf16(
                    false, fa[mi].b, false, fb[ni].b,
                    (short)0, acc[mi][ni], false, false);
    }

    // Epilogue: BN + ReLU, store f32. C/D layout: VGPR r -> M = r + 8*half,
    // N = lane&15  => 16 lanes store a contiguous 64B run per row.
#pragma unroll
    for (int mi = 0; mi < 4; ++mi) {
#pragma unroll
        for (int r = 0; r < 8; ++r) {
            int row = m_base + mi * 16 + half * 8 + r;
            float sc = scale[row], sh = shift[row];
#pragma unroll
            for (int ni = 0; ni < 2; ++ni) {
                int col = n_base + ni * 16 + l16;
                float v = acc[mi][ni][r] * sc + sh;
                out[(size_t)row * HW_IN + col] = v > 0.f ? v : 0.f;
            }
        }
    }
}

// ---------------------------------------------------------------------------
extern "C" void kernel_launch(void* const* d_in, const int* in_sizes, int n_in,
                              void* d_out, int out_size, void* d_ws, size_t ws_size,
                              hipStream_t stream) {
    const float* x        = (const float*)d_in[0];
    const float* dw_w     = (const float*)d_in[1];
    const float* dw_gamma = (const float*)d_in[2];
    const float* dw_beta  = (const float*)d_in[3];
    const float* dw_mean  = (const float*)d_in[4];
    const float* dw_var   = (const float*)d_in[5];
    const float* pw_w     = (const float*)d_in[6];
    const float* pw_gamma = (const float*)d_in[7];
    const float* pw_beta  = (const float*)d_in[8];
    const float* pw_mean  = (const float*)d_in[9];
    const float* pw_var   = (const float*)d_in[10];
    float* out = (float*)d_out;

    // Workspace layout (all 16B aligned):
    //   [0)            wb        : 512*256 bf16  = 262144 B
    //   [262144)       dw_scale  : 256 f32
    //   [263168)       dw_shift  : 256 f32
    //   [264192)       pw_scale  : 512 f32
    //   [266240)       pw_shift  : 512 f32
    //   [268288)       y bf16    : 50176*256*2   = 25690112 B
    // total ~25.96 MB (y stays resident in the 192 MB L2)
    char* ws = (char*)d_ws;
    unsigned short* wb = (unsigned short*)ws;
    float* dw_scale = (float*)(ws + 262144);
    float* dw_shift = dw_scale + C_IN;
    float* pw_scale = dw_shift + C_IN;
    float* pw_shift = pw_scale + C_OUT;
    unsigned short* y = (unsigned short*)(ws + 268288);

    prep_kernel<<<dim3((C_OUT * C_IN + 255) / 256), dim3(256), 0, stream>>>(
        pw_w, dw_gamma, dw_beta, dw_mean, dw_var,
        pw_gamma, pw_beta, pw_mean, pw_var,
        wb, dw_scale, dw_shift, pw_scale, pw_shift);

    dw_kernel<<<dim3(H_IMG, C_IN), dim3(W_IMG), 0, stream>>>(
        x, dw_w, dw_scale, dw_shift, y);

    // N = 50176 = 392 * 128, M = 512 = 4 * 128
    pw_gemm_kernel<<<dim3(HW_IN / 128, C_OUT / 128), dim3(256), 0, stream>>>(
        wb, y, pw_scale, pw_shift, out);
}
